// Transition_17875653886346
// MI455X (gfx1250) — compile-verified, matrix operand order checked by reading, not run on hardware
//
#include <hip/hip_runtime.h>
#include <hip/hip_bf16.h>

typedef __attribute__((ext_vector_type(16))) _Float16     v16h;
typedef __attribute__((ext_vector_type(4)))  _Float16     v4h;
typedef __attribute__((ext_vector_type(8)))  float        v8f;
typedef __attribute__((ext_vector_type(4)))  unsigned int v4u;
typedef __attribute__((ext_vector_type(8)))  unsigned int v8u;

#define HDIM 1024
#define INDIM 16

__device__ __forceinline__ v8f wmma_f16(v16h a, v16h b, v8f c) {
    return __builtin_amdgcn_wmma_f32_16x16x32_f16(false, a, false, b,
                                                  (short)0, c, false, false);
}

// ReLU in one VALU op: v_med3_f32(x, 0, +inf) clamps to [0, inf).
__device__ __forceinline__ float relu1(float x) {
    return __builtin_amdgcn_fmed3f(x, 0.0f, __builtin_inff());
}

// Butterfly sum across the 16 lanes of each half-wave via ds_swizzle XOR swaps
// (offset[14:10]=xor, [9:5]=or, [4:0]=and; and=0x1f keeps full 32-lane group).
__device__ __forceinline__ float half16_reduce(float x) {
    x += __int_as_float(__builtin_amdgcn_ds_swizzle(__float_as_int(x), 0x041f)); // XOR 1
    x += __int_as_float(__builtin_amdgcn_ds_swizzle(__float_as_int(x), 0x081f)); // XOR 2
    x += __int_as_float(__builtin_amdgcn_ds_swizzle(__float_as_int(x), 0x101f)); // XOR 4
    x += __int_as_float(__builtin_amdgcn_ds_swizzle(__float_as_int(x), 0x201f)); // XOR 8
    return x;
}

__global__ __launch_bounds__(256, 1)
void transition_fused_kernel(const float* __restrict__ x0,
                             const float* __restrict__ sa,
                             const float* __restrict__ dt_p,
                             const float* __restrict__ unorm_p,
                             const float* __restrict__ pnorm_p,
                             const float* __restrict__ W1l, const float* __restrict__ b1l,
                             const float* __restrict__ W2l, const float* __restrict__ b2l,
                             const float* __restrict__ W1a, const float* __restrict__ b1a,
                             const float* __restrict__ W2a, const float* __restrict__ b2a,
                             float* __restrict__ out, int Btot,
                             int ntiles /* = HDIM/16, runtime to keep loop rolled */)
{
    const int lane = threadIdx.x & 31;
    const int wave = threadIdx.x >> 5;
    int tileM = blockIdx.x * 8 + wave;           // 16-row batch tile per wave
    const int maxTile = Btot / 16 - 1;
    if (tileM > maxTile) tileM = maxTile;        // clamp (no return: barriers below)
    const int row0 = tileM * 16;

    // LDS: W1 as f16 (32KB), bias pre-packed {f16,0} u32 + zero shadow row (8KB),
    // W2 as f32 (12KB), per-wave dot results (3KB).  Total ~55KB.
    __shared__ _Float16     sW1h[HDIM * INDIM];
    __shared__ unsigned int sB1u[2 * HDIM];
    __shared__ float        sW2[3 * HDIM];
    __shared__ float        s_dot[8][16][6];

    const float dt = *dt_p;

    // ---- A matrix: 16x32 f16.  K=0..15 real, K=16 = constant 1.0 (bias lane),
    // K=17..31 zero.  Layout: lanes 0-15 hold M=lane, elems 0..7 = K0..7,
    // elems 8..15 = K16..23; lanes 16-31 hold M=lane-16, elems 0..7 = K8..15,
    // elems 8..15 = K24..31.
    const int  arow = row0 + (lane & 15);
    const int  koff = (lane < 16) ? 0 : 8;
    const float* ap = sa + arow * INDIM + koff;
    v16h A;
    #pragma unroll
    for (int i = 0; i < 8; ++i)  A[i] = (_Float16)ap[i];
    #pragma unroll
    for (int i = 8; i < 16; ++i) A[i] = (_Float16)0.0f;
    A[8] = (lane < 16) ? (_Float16)1.0f : (_Float16)0.0f;   // K=16 ones column

    const int ncol = lane & 15;              // this lane's N column in the C tile
    const int bsel = (lane >> 4) * HDIM;     // lanes>=16 read the zero shadow row

    // ---- Stage one MLP's weights into LDS (block-cooperative).
    auto stage = [&](const float* __restrict__ W1, const float* __restrict__ b1,
                     const float* __restrict__ W2) {
        const float4* w1v = (const float4*)W1;
        v4h* s1v = (v4h*)sW1h;
        for (int i = threadIdx.x; i < HDIM * INDIM / 4; i += 256) {
            float4 f = w1v[i];
            v4h h; h[0] = (_Float16)f.x; h[1] = (_Float16)f.y;
                   h[2] = (_Float16)f.z; h[3] = (_Float16)f.w;
            s1v[i] = h;
        }
        for (int i = threadIdx.x; i < HDIM; i += 256) {
            unsigned short hb = __builtin_bit_cast(unsigned short, (_Float16)b1[i]);
            sB1u[i]        = (unsigned int)hb;   // {f16 bias, f16 zero}
            sB1u[HDIM + i] = 0u;                 // shadow row for hi half-wave
        }
        for (int i = threadIdx.x; i < 3 * HDIM; i += 256) sW2[i] = W2[i];
    };

    // ---- Fused MLP: layer1 via WMMA over ntiles N-tiles (bias folded into
    // K=16), layer2 folded in as a 3-wide FMA reduction.  Software-pipelined
    // one tile deep: the ReLU/FMA VALU reads the *previous* iteration's WMMA D
    // registers, so the matrix pipe co-executes with the VALU work.
    auto run_mlp = [&](const float* __restrict__ b2, float norm, int off) {
        float acc[24];
        #pragma unroll
        for (int i = 0; i < 24; ++i) acc[i] = 0.0f;

        auto loadB = [&](int t) -> v16h {
            const int n = t * 16 + ncol;
            v4u lo = *(const v4u*)(sW1h + n * INDIM + koff);   // 8 f16, 16B aligned
            v8u Bu;
            Bu[0] = lo[0]; Bu[1] = lo[1]; Bu[2] = lo[2]; Bu[3] = lo[3];
            Bu[4] = sB1u[bsel + n];       // {bias,0} for lo half, {0,0} for hi half
            Bu[5] = 0u; Bu[6] = 0u; Bu[7] = 0u;                // loop-invariant zeros
            return __builtin_bit_cast(v16h, Bu);
        };

        // prologue: tile 0
        v8f  c_pipe = wmma_f16(A, loadB(0), v8f{});
        float w0 = sW2[ncol], w1 = sW2[HDIM + ncol], w2 = sW2[2 * HDIM + ncol];

        #pragma unroll 2
        for (int t = 1; t < ntiles; ++t) {
            v8f c_next = wmma_f16(A, loadB(t), v8f{});
            const int n = t * 16 + ncol;
            float n0 = sW2[n], n1 = sW2[HDIM + n], n2 = sW2[2 * HDIM + n];
            #pragma unroll
            for (int v = 0; v < 8; ++v) {
                float h = relu1(c_pipe[v]);         // bias already in via GEMM
                acc[v * 3 + 0] = fmaf(h, w0, acc[v * 3 + 0]);
                acc[v * 3 + 1] = fmaf(h, w1, acc[v * 3 + 1]);
                acc[v * 3 + 2] = fmaf(h, w2, acc[v * 3 + 2]);
            }
            c_pipe = c_next; w0 = n0; w1 = n1; w2 = n2;
        }
        #pragma unroll
        for (int v = 0; v < 8; ++v) {        // epilogue: last tile
            float h = relu1(c_pipe[v]);
            acc[v * 3 + 0] = fmaf(h, w0, acc[v * 3 + 0]);
            acc[v * 3 + 1] = fmaf(h, w1, acc[v * 3 + 1]);
            acc[v * 3 + 2] = fmaf(h, w2, acc[v * 3 + 2]);
        }

        #pragma unroll
        for (int i = 0; i < 24; ++i) acc[i] = half16_reduce(acc[i]);

        // Lane 0 holds rows 0..7 (M=v), lane 16 holds rows 8..15 (M=v+8).
        if ((lane & 15) == 0) {
            const int rbase = (lane >> 4) * 8;
            #pragma unroll
            for (int v = 0; v < 8; ++v)
                #pragma unroll
                for (int j = 0; j < 3; ++j)
                    s_dot[wave][rbase + v][off + j] = (acc[v * 3 + j] + b2[j]) * norm;
        }
    };

    stage(W1l, b1l, W2l);
    __syncthreads();
    run_mlp(b2l, *unorm_p, 0);               // uvw_dot
    __syncthreads();                          // all waves done with linear weights
    stage(W1a, b1a, W2a);
    __syncthreads();
    run_mlp(b2a, *pnorm_p, 3);               // pqr_dot
    __syncthreads();

    // ---- Elementwise epilogue: 16 rows per wave, one row per lane 0..15.
    if (lane < 16) {
        const int g = row0 + lane;
        const float* s = sa + g * INDIM;
        const long long NB3 = (long long)Btot * 3;

        float z0 = asinf(s[0]);
        float z1 = asinf(s[1]);
        float z2 = asinf(s[2]);

        float u  = s[6]  + s_dot[wave][lane][0] * dt;
        float vv = s[7]  + s_dot[wave][lane][1] * dt;
        float w  = s[8]  + s_dot[wave][lane][2] * dt;
        float p  = s[9]  + s_dot[wave][lane][3] * dt;
        float q  = s[10] + s_dot[wave][lane][4] * dt;
        float r  = s[11] + s_dot[wave][lane][5] * dt;

        float s1, c1, s2, c2, s3, c3;
        __sincosf(z0, &s1, &c1);
        __sincosf(z1, &s2, &c2);
        __sincosf(z2, &s3, &c3);
        const float t2   = s2 / c2;          // tan(z1)
        const float sec2 = 1.0f / c2;

        float xd = c2 * c3 * u + (s1 * s2 * c3 - c3 * s3) * vv + (c1 * s2 * c3 + s1 * s3) * w;
        float yd = c2 * s3 * u + (s1 * s2 * s3 + c1 * c3) * vv + (c1 * s2 * s3 - s1 * c3) * w;
        float zd = -s2 * u + s1 * c2 * vv + c1 * c2 * w;
        float e0 = p + s1 * t2 * q + c1 * t2 * r;
        float e1 = c1 * q - s1 * r;
        float e2 = s1 * sec2 * q + c1 * sec2 * r;

        out[0 * NB3 + g * 3 + 0] = x0[g * 3 + 0] + xd * dt;
        out[0 * NB3 + g * 3 + 1] = x0[g * 3 + 1] + yd * dt;
        out[0 * NB3 + g * 3 + 2] = x0[g * 3 + 2] + zd * dt;

        out[1 * NB3 + g * 3 + 0] = z0 + e0 * dt;
        out[1 * NB3 + g * 3 + 1] = z1 + e1 * dt;
        out[1 * NB3 + g * 3 + 2] = z2 + e2 * dt;

        out[2 * NB3 + g * 3 + 0] = u;
        out[2 * NB3 + g * 3 + 1] = vv;
        out[2 * NB3 + g * 3 + 2] = w;

        out[3 * NB3 + g * 3 + 0] = p;
        out[3 * NB3 + g * 3 + 1] = q;
        out[3 * NB3 + g * 3 + 2] = r;
    }
}

extern "C" void kernel_launch(void* const* d_in, const int* in_sizes, int n_in,
                              void* d_out, int out_size, void* d_ws, size_t ws_size,
                              hipStream_t stream) {
    (void)n_in; (void)d_ws; (void)ws_size; (void)out_size;
    const float* x0  = (const float*)d_in[0];
    const float* sa  = (const float*)d_in[1];
    const float* dt  = (const float*)d_in[2];
    const float* un  = (const float*)d_in[3];
    const float* pn  = (const float*)d_in[4];
    const float* W1l = (const float*)d_in[5];
    const float* b1l = (const float*)d_in[6];
    const float* W2l = (const float*)d_in[7];
    const float* b2l = (const float*)d_in[8];
    const float* W1a = (const float*)d_in[9];
    const float* b1a = (const float*)d_in[10];
    const float* W2a = (const float*)d_in[11];
    const float* b2a = (const float*)d_in[12];

    const int Btot   = in_sizes[1] / INDIM;      // state_action is (B, 16)
    const int ntiles = HDIM / 16;                // 64, runtime to keep loop rolled
    dim3 block(256);                             // 8 wave32 per block
    dim3 grid((Btot + 127) / 128);               // 128 rows per block

    transition_fused_kernel<<<grid, block, 0, stream>>>(
        x0, sa, dt, un, pn, W1l, b1l, W2l, b2l, W1a, b1a, W2a, b2a,
        (float*)d_out, Btot, ntiles);
}
